// Decoder_80350248173764
// MI455X (gfx1250) — matrix operand order repaired and checked by reading, not compiled, not run on hardware
//
#include <hip/hip_runtime.h>
#include <math.h>

// ---------------------------------------------------------------------------
// LSTM decoder for MI455X (gfx1250, wave32, WMMA + Tensor Data Mover).
//   1) prep:  gather embeddings -> bf16, convert weights -> bf16, init state.
//   2) gemm:  xg = xbf @ W_ih^T + (b_ih+b_hh)      (TDM double-buffer + WMMA)
//   3) lstm:  persistent kernel, 128 steps, WMMA for h@W_hh^T, grid barrier.
//   4) gemm:  out = hs @ W_fc^T + b_fc             (TDM double-buffer + WMMA)
// ---------------------------------------------------------------------------

typedef unsigned short u16;
typedef __attribute__((ext_vector_type(16))) __bf16 v16bf;
typedef __attribute__((ext_vector_type(8)))  __bf16 v8bf;
typedef __attribute__((ext_vector_type(8)))  float  v8f;
typedef __attribute__((ext_vector_type(4)))  unsigned int v4u;
typedef __attribute__((ext_vector_type(4)))  int v4i;
typedef __attribute__((ext_vector_type(8)))  int v8i;

#define Bsz 16
#define Ssz 128
#define Esz 512
#define Hsz 1024
#define G4H 4096
#define Vsz 32000
#define LSTM_WGS 32

// round-to-nearest-even f32 -> bf16
__device__ __forceinline__ u16 f2bf(float f) {
  union { float f; unsigned u; } v; v.f = f;
  unsigned r = v.u + 0x7FFFu + ((v.u >> 16) & 1u);
  return (u16)(r >> 16);
}

// A-fragment (16x32 MxK bf16, row-major, leading dim lda)
__device__ __forceinline__ v16bf load_a_frag(const u16* A, int lda, int row0, int k0) {
  int lane = threadIdx.x & 31;
  int row  = row0 + (lane & 15);
  int k    = k0 + ((lane >> 4) << 4);
  const __bf16* p = reinterpret_cast<const __bf16*>(A + (size_t)row * lda + k);
  v8bf lo = *reinterpret_cast<const v8bf*>(p);
  v8bf hi = *reinterpret_cast<const v8bf*>(p + 8);
  v16bf f;
#pragma unroll
  for (int i = 0; i < 8; ++i) { f[i] = lo[i]; f[i + 8] = hi[i]; }
  return f;
}

// B-fragment (32x16 KxN) from weights stored [N,K] row-major: one 32B load/lane
__device__ __forceinline__ v16bf load_b_frag(const u16* W, int ldb, int n0, int k0) {
  int lane = threadIdx.x & 31;
  const u16* p = W + (size_t)(n0 + (lane & 15)) * ldb + k0 + ((lane >> 4) << 4);
  return *reinterpret_cast<const v16bf*>(p);
}

// ---------------- Tensor Data Mover: 2D tile (tile0 x tile1) ----------------
// Loads a [tile1 rows x tile0 elems] bf16 tile (row-major, row stride = stride0
// elements) from global memory into LDS at byte offset lds_off. Tracked by
// TENSORcnt.  D# layout per CDNA5 ISA ch.8 (group0 128b, group1 256b).
__device__ __forceinline__ void tdm_load_2d(const u16* gsrc, unsigned lds_off,
                                            int dim0, int dim1, int tile0,
                                            int tile1, int stride0) {
  unsigned long long ga = (unsigned long long)(uintptr_t)gsrc;
  v4u g0;
  g0[0] = 1u;                                             // count=1 (valid)
  g0[1] = lds_off;                                        // lds_addr
  g0[2] = (unsigned)(ga & 0xFFFFFFFFu);                   // global_addr lo
  g0[3] = (unsigned)((ga >> 32) & 0x01FFFFFFu) | (2u << 30); // addr hi | type=2
  v8i g1;
  g1[0] = 1 << 16;                                        // data_size=1 (2B)
  g1[1] = (int)(((unsigned)dim0 & 0xFFFFu) << 16);        // tensor_dim0 lo
  g1[2] = (int)((((unsigned)dim0 >> 16) & 0xFFFFu) |
                (((unsigned)dim1 & 0xFFFFu) << 16));      // dim0 hi | dim1 lo
  g1[3] = (int)((((unsigned)dim1 >> 16) & 0xFFFFu) |
                (((unsigned)tile0 & 0xFFFFu) << 16));     // dim1 hi | tile_dim0
  g1[4] = (int)((unsigned)tile1 & 0xFFFFu);               // tile_dim1 (dim2=0)
  g1[5] = (int)(unsigned)stride0;                         // dim0_stride lo
  g1[6] = 0;                                              // stride hi / dim1_stride
  g1[7] = 0;
  v4i z = {};
#if defined(__clang_major__) && __clang_major__ >= 23
  v8i z8 = {};
  __builtin_amdgcn_tensor_load_to_lds(g0, g1, z, z, z8, 0);
#else
  __builtin_amdgcn_tensor_load_to_lds(g0, g1, z, z, 0);
#endif
}

__device__ __forceinline__ unsigned lds_off_of(const void* p) {
  // generic pointers to LDS carry the byte offset in the low 32 bits
  return (unsigned)(uintptr_t)p;
}

// -------------------------- prep / conversion ------------------------------
__global__ __launch_bounds__(256)
void prep_kernel(const int* __restrict__ tgt, const float* __restrict__ h0,
                 const float* __restrict__ c0, const float* __restrict__ emb,
                 const float* __restrict__ Wih, const float* __restrict__ Whh,
                 const float* __restrict__ Wfc,
                 u16* __restrict__ xbf, u16* __restrict__ wihb,
                 u16* __restrict__ whhb, u16* __restrict__ wfcb,
                 u16* __restrict__ hbf, float* __restrict__ cst,
                 int* __restrict__ bar) {
  size_t tid = (size_t)blockIdx.x * blockDim.x + threadIdx.x;
  size_t nth = (size_t)gridDim.x * blockDim.x;
  if (tid == 0) { bar[0] = 0; bar[64] = 0; }
  for (size_t i = tid; i < (size_t)Bsz * Ssz * Esz; i += nth) {
    size_t row = i >> 9; int col = (int)(i & 511);
    xbf[i] = f2bf(emb[(size_t)tgt[row] * Esz + col]);
  }
  for (size_t i = tid; i < (size_t)G4H * Esz; i += nth) wihb[i] = f2bf(Wih[i]);
  for (size_t i = tid; i < (size_t)G4H * Hsz; i += nth) whhb[i] = f2bf(Whh[i]);
  for (size_t i = tid; i < (size_t)Vsz * Hsz; i += nth) wfcb[i] = f2bf(Wfc[i]);
  for (size_t i = tid; i < (size_t)Bsz * Hsz; i += nth) {
    hbf[i] = f2bf(h0[i]);
    cst[i] = c0[i];
  }
}

// ---------------- bf16 WMMA GEMM with TDM double buffering -----------------
// C[M,N] = A[M,K] @ B[N,K]^T + bias0 (+ bias1).  WG tile 128x128, 8 waves,
// wave tile 32x64.  Wave 0 DMAs the next 128x32 A/B K-slabs into LDS with
// tensor_load_to_lds while all waves run WMMA on the current slab.
__global__ __launch_bounds__(256)
void gemm_bf16_wmma(const u16* __restrict__ A, const u16* __restrict__ Bw,
                    const float* __restrict__ bias0, const float* __restrict__ bias1,
                    float* __restrict__ C, int N, int K) {
  __shared__ __align__(64) u16 sA[2][128 * 32];   // 16 KB
  __shared__ __align__(64) u16 sB[2][128 * 32];   // 16 KB
  int wave = threadIdx.x >> 5;
  int lane = threadIdx.x & 31;
  int rowBase = blockIdx.y * 128;
  int colBase = blockIdx.x * 128;
  int wm = (wave & 3) * 32;
  int wn = (wave >> 2) * 64;
  // uniform (scalar-branch) predicate: TDM ignores EXEC, so gate by wave id
  bool issuer = (__builtin_amdgcn_readfirstlane((int)threadIdx.x) == 0);

  v8f acc[2][4];
#pragma unroll
  for (int i = 0; i < 2; ++i)
#pragma unroll
    for (int j = 0; j < 4; ++j) { v8f z = {}; acc[i][j] = z; }

  if (issuer) {
    tdm_load_2d(A + (size_t)rowBase * K, lds_off_of(&sA[0][0]), K, 131072, 32, 128, K);
    tdm_load_2d(Bw + (size_t)colBase * K, lds_off_of(&sB[0][0]), K, 131072, 32, 128, K);
  }

  int nIter = K >> 5;
  for (int it = 0; it < nIter; ++it) {
    int cur = it & 1;
    if (issuer) __builtin_amdgcn_s_wait_tensorcnt(0);
    __syncthreads();   // slab `cur` ready; all waves done reading slab cur^1
    if (issuer && (it + 1) < nIter) {
      int k0 = (it + 1) << 5;
      tdm_load_2d(A + (size_t)rowBase * K + k0, lds_off_of(&sA[cur ^ 1][0]),
                  K, 131072, 32, 128, K);
      tdm_load_2d(Bw + (size_t)colBase * K + k0, lds_off_of(&sB[cur ^ 1][0]),
                  K, 131072, 32, 128, K);
    }
    const u16* a = &sA[cur][0];
    const u16* b = &sB[cur][0];
    v16bf a0 = load_a_frag(a, 32, wm, 0);
    v16bf a1 = load_a_frag(a, 32, wm + 16, 0);
#pragma unroll
    for (int j = 0; j < 4; ++j) {
      v16bf bf = load_b_frag(b, 32, wn + 16 * j, 0);
      acc[0][j] = __builtin_amdgcn_wmma_f32_16x16x32_bf16(false, a0, false, bf,
                                                          (short)0, acc[0][j], false, false);
      acc[1][j] = __builtin_amdgcn_wmma_f32_16x16x32_bf16(false, a1, false, bf,
                                                          (short)0, acc[1][j], false, false);
    }
  }

  int nlane = lane & 15;
  int mlane = (lane >> 4) << 3;
#pragma unroll
  for (int i = 0; i < 2; ++i) {
#pragma unroll
    for (int j = 0; j < 4; ++j) {
      int col = colBase + wn + 16 * j + nlane;
      float bv = bias0[col];
      if (bias1) bv += bias1[col];
#pragma unroll
      for (int r = 0; r < 8; ++r) {
        int row = rowBase + wm + 16 * i + mlane + r;
        C[(size_t)row * N + col] = acc[i][j][r] + bv;
      }
    }
  }
}

// -------------------------- grid barrier -----------------------------------
__device__ __forceinline__ void grid_barrier(int* cnt, volatile int* gen, int& lg) {
  __syncthreads();
  if (threadIdx.x == 0) {
    __threadfence();
    if (atomicAdd(cnt, 1) == LSTM_WGS - 1) {
      atomicExch(cnt, 0);
      __threadfence();
      atomicAdd((int*)gen, 1);
    } else {
      while (*gen <= lg) __builtin_amdgcn_s_sleep(1);
      __threadfence();
    }
  }
  lg++;
  __syncthreads();
}

// -------------------------- persistent LSTM --------------------------------
// 32 WGs x 8 waves = 256 waves; each wave owns one 16-wide gate-column tile
// (256 tiles cover 4H).  M=16=B is exactly one WMMA tile row.
__global__ __launch_bounds__(256)
void lstm_persistent(const float* __restrict__ xg,      // [B*S, 4H]
                     const u16* __restrict__ Whh,       // [4H, H] bf16
                     u16* __restrict__ hbf,             // [B, H]  bf16 state
                     float* __restrict__ cst,           // [B, H]  f32 state
                     float* __restrict__ gbuf,          // [B, 4H] f32 gates
                     u16* __restrict__ hsbf,            // [B*S, H] bf16 outputs
                     int* __restrict__ bar) {
  int wave = threadIdx.x >> 5;
  int lane = threadIdx.x & 31;
  int n0 = (blockIdx.x * 8 + wave) * 16;
  int* cnt = bar;
  volatile int* gen = bar + 64;
  int lg = 0;

  for (int t = 0; t < Ssz; ++t) {
    v8f acc = {};
    for (int k0 = 0; k0 < Hsz; k0 += 32) {
      if (k0 + 32 < Hsz)  // keep the L2-resident W_hh rows streaming
        __builtin_prefetch((const void*)(Whh + (size_t)(n0 + (lane & 15)) * Hsz + k0 + 32), 0, 0);
      v16bf a = load_a_frag(hbf, Hsz, 0, k0);
      v16bf b = load_b_frag(Whh, Hsz, n0, k0);
      acc = __builtin_amdgcn_wmma_f32_16x16x32_bf16(false, a, false, b,
                                                    (short)0, acc, false, false);
    }
    int n = n0 + (lane & 15);
    int mb = (lane >> 4) << 3;
#pragma unroll
    for (int r = 0; r < 8; ++r) {
      int m = mb + r;
      gbuf[m * G4H + n] = acc[r] + xg[(size_t)(m * Ssz + t) * G4H + n];
    }
    grid_barrier(cnt, gen, lg);

    for (int u = blockIdx.x * 256 + threadIdx.x; u < Bsz * Hsz; u += LSTM_WGS * 256) {
      int b = u >> 10, j = u & (Hsz - 1);
      const float* gb = gbuf + b * G4H;
      float iv = 1.f / (1.f + __expf(-gb[j]));
      float fv = 1.f / (1.f + __expf(-gb[Hsz + j]));
      float gv = tanhf(gb[2 * Hsz + j]);
      float ov = 1.f / (1.f + __expf(-gb[3 * Hsz + j]));
      float cv = fv * cst[u] + iv * gv;
      cst[u] = cv;
      u16 hb = f2bf(ov * tanhf(cv));
      hbf[u] = hb;
      hsbf[(size_t)(b * Ssz + t) * Hsz + j] = hb;
    }
    grid_barrier(cnt, gen, lg);
  }
}

// ---------------------------------------------------------------------------
extern "C" void kernel_launch(void* const* d_in, const int* in_sizes, int n_in,
                              void* d_out, int out_size, void* d_ws, size_t ws_size,
                              hipStream_t stream) {
  const int*   tgt = (const int*)d_in[0];
  const float* h0  = (const float*)d_in[1];
  const float* c0  = (const float*)d_in[2];
  const float* emb = (const float*)d_in[3];
  const float* Wih = (const float*)d_in[4];
  const float* Whh = (const float*)d_in[5];
  const float* bih = (const float*)d_in[6];
  const float* bhh = (const float*)d_in[7];
  const float* Wfc = (const float*)d_in[8];
  const float* bfc = (const float*)d_in[9];
  float* out = (float*)d_out;

  char* w = (char*)d_ws;
  auto carve = [&](size_t bytes) {
    char* p = w;
    w += (bytes + 255) & ~(size_t)255;
    return p;
  };
  u16*   xbf  = (u16*)  carve((size_t)Bsz * Ssz * Esz * 2);
  u16*   wihb = (u16*)  carve((size_t)G4H * Esz * 2);
  u16*   whhb = (u16*)  carve((size_t)G4H * Hsz * 2);
  u16*   wfcb = (u16*)  carve((size_t)Vsz * Hsz * 2);
  float* xg   = (float*)carve((size_t)Bsz * Ssz * G4H * 4);
  u16*   hbf  = (u16*)  carve((size_t)Bsz * Hsz * 2);
  float* cst  = (float*)carve((size_t)Bsz * Hsz * 4);
  float* gbuf = (float*)carve((size_t)Bsz * G4H * 4);
  u16*   hsbf = (u16*)  carve((size_t)Bsz * Ssz * Hsz * 2);
  int*   bar  = (int*)  carve(512);

  prep_kernel<<<4096, 256, 0, stream>>>(tgt, h0, c0, emb, Wih, Whh, Wfc,
                                        xbf, wihb, whhb, wfcb, hbf, cst, bar);

  dim3 g1(G4H / 128, (Bsz * Ssz) / 128);   // 32 x 16
  gemm_bf16_wmma<<<g1, 256, 0, stream>>>(xbf, wihb, bih, bhh, xg, G4H, Esz);

  lstm_persistent<<<LSTM_WGS, 256, 0, stream>>>(xg, whhb, hbf, cst, gbuf, hsbf, bar);

  dim3 g2(Vsz / 128, (Bsz * Ssz) / 128);   // 250 x 16
  gemm_bf16_wmma<<<g2, 256, 0, stream>>>(hsbf, wfcb, bfc, nullptr, out, Vsz, Hsz);
}
